// Controller_10325101379613
// MI455X (gfx1250) — matrix operand order repaired and checked by reading, not compiled, not run on hardware
//
#include <hip/hip_runtime.h>
#include <math.h>

#define HDIM 512
#define ZDIM 512
#define EDIM 512
#define BSZ  32

typedef __attribute__((ext_vector_type(16))) __bf16 v16bf;
typedef __attribute__((ext_vector_type(8)))  float  v8f;

// ---------------- WMMA fragment loaders (CDNA5 16-bit layouts, wave32) -------------
// A: 16x32 (MxK). lanes 0-15: M=lane, K={0..7,16..23}; lanes 16-31: M=lane-16, K={8..15,24..31}
__device__ __forceinline__ v16bf load_A_frag(const __bf16* A, int lda, int m0, int k0, int lane) {
    int m = m0 + (lane & 15);
    int kh = lane >> 4;                 // 0 or 1
    const __bf16* p0 = A + (long)m * lda + k0 + kh * 8;
    union { v16bf v; uint4 q[2]; } u;
    u.q[0] = *(const uint4*)p0;         // K = k0+kh*8 .. +7
    u.q[1] = *(const uint4*)(p0 + 16);  // K = k0+kh*8+16 .. +7
    return u.v;
}
// B: 32x16 (KxN), B[k][n] = W[n0+n][k] (W row-major NxK). lanes 0-15: N=lane, K={0..7,16..23};
// lanes 16-31: N=lane-16, K={8..15,24..31}.
__device__ __forceinline__ v16bf load_B_frag(const __bf16* W, int ldw, int n0, int k0, int lane) {
    int n = n0 + (lane & 15);
    int kh = lane >> 4;
    const __bf16* p = W + (long)n * ldw + k0 + kh * 8;
    union { v16bf v; uint4 q[2]; } u;
    u.q[0] = *(const uint4*)p;
    u.q[1] = *(const uint4*)(p + 16);
    return u.v;
}

__device__ __forceinline__ float sigf(float x) { return 1.0f / (1.0f + __expf(-x)); }

// ---------------- conversion / init kernels ---------------------------------------
__global__ void k_f32_to_bf16(const float* __restrict__ in, __bf16* __restrict__ out, long n) {
    long i = (long)blockIdx.x * blockDim.x + threadIdx.x;
    long stride = (long)gridDim.x * blockDim.x;
    for (; i < n; i += stride) out[i] = (__bf16)in[i];
}
__global__ void k_bias_sum(const float* bi, const float* bh, float* bs, int n) {
    int i = blockIdx.x * blockDim.x + threadIdx.x;
    if (i < n) bs[i] = bi[i] + bh[i];
}
__global__ void k_init_state(const float* init_z, __bf16* zbf, __bf16* h0, __bf16* h1, float* c) {
    int i = blockIdx.x * blockDim.x + threadIdx.x;   // BSZ*HDIM
    if (i < BSZ * HDIM) {
        zbf[i] = (__bf16)init_z[i & (ZDIM - 1)];
        h0[i]  = (__bf16)0.0f;
        h1[i]  = (__bf16)0.0f;
        c[i]   = 0.0f;
    }
}

// ---------------- LSTM gates + cell update -----------------------------------------
// grid = HDIM/16 blocks (column slice jb), 256 threads = 8 waves.
// wave w: mt = w&1 (rows), g = w>>1 (gate 0..3). Tile N0 = g*HDIM + jb*16, K = 512(z) + 512(h).
__global__ __launch_bounds__(256) void k_lstm(
    const __bf16* __restrict__ zbf, const __bf16* __restrict__ hin,
    const __bf16* __restrict__ Wih, const __bf16* __restrict__ Whh,
    const float* __restrict__ bsum, float* __restrict__ c, __bf16* __restrict__ hout) {
    int jb = blockIdx.x;
    int lane = threadIdx.x & 31;
    int w = threadIdx.x >> 5;
    int mt = w & 1;
    int g  = w >> 1;
    int m0 = mt * 16;
    int N0 = g * HDIM + jb * 16;

    v8f acc = {};
    #pragma unroll 4
    for (int k = 0; k < ZDIM; k += 32) {
        v16bf a = load_A_frag(zbf, ZDIM, m0, k, lane);
        v16bf b = load_B_frag(Wih, ZDIM, N0, k, lane);
        acc = __builtin_amdgcn_wmma_f32_16x16x32_bf16(false, a, false, b, (short)0, acc, false, false);
    }
    #pragma unroll 4
    for (int k = 0; k < HDIM; k += 32) {
        v16bf a = load_A_frag(hin, HDIM, m0, k, lane);
        v16bf b = load_B_frag(Whh, HDIM, N0, k, lane);
        acc = __builtin_amdgcn_wmma_f32_16x16x32_bf16(false, a, false, b, (short)0, acc, false, false);
    }
    int n  = lane & 15;
    int rh = lane >> 4;
    float bias = bsum[N0 + n];

    __shared__ float gl[4][BSZ][16];
    #pragma unroll
    for (int v = 0; v < 8; v++)
        gl[g][m0 + v + (rh << 3)][n] = acc[v] + bias;
    __syncthreads();

    // pointwise LSTM cell: 32 rows x 16 cols = 512 elems, 256 threads x 2
    for (int e = threadIdx.x; e < BSZ * 16; e += 256) {
        int row = e >> 4;
        int nc  = e & 15;
        int col = jb * 16 + nc;
        float iv = gl[0][row][nc], fv = gl[1][row][nc];
        float gv = gl[2][row][nc], ov = gl[3][row][nc];
        long idx = (long)row * HDIM + col;
        float cn = sigf(fv) * c[idx] + sigf(iv) * tanhf(gv);
        float hn = sigf(ov) * tanhf(cn);
        c[idx] = cn;
        hout[idx] = (__bf16)hn;
    }
}

// ---------------- generic 32xN = (32xK)·(NxK)^T WMMA GEMM --------------------------
// grid = N/64 blocks, 256 threads (8 waves, 1 tile each). Optional f32 + bf16 outputs.
__global__ __launch_bounds__(256) void k_gemm32(
    const __bf16* __restrict__ A, const __bf16* __restrict__ W, int K,
    float* __restrict__ outf, long ldof, __bf16* __restrict__ outb, int ldob) {
    int lane = threadIdx.x & 31;
    int wid  = (blockIdx.x << 3) + (threadIdx.x >> 5);
    int m0 = (wid & 1) * 16;
    int n0 = (wid >> 1) * 16;

    v8f acc = {};
    #pragma unroll 4
    for (int k = 0; k < K; k += 32) {
        v16bf a = load_A_frag(A, K, m0, k, lane);
        v16bf b = load_B_frag(W, K, n0, k, lane);
        acc = __builtin_amdgcn_wmma_f32_16x16x32_bf16(false, a, false, b, (short)0, acc, false, false);
    }
    int n  = n0 + (lane & 15);
    int rh = lane >> 4;
    #pragma unroll
    for (int v = 0; v < 8; v++) {
        int m = m0 + v + (rh << 3);
        if (outf) outf[(long)m * ldof + n] = acc[v];
        if (outb) outb[(long)m * ldob + n] = (__bf16)acc[v];
    }
}

// ---------------- fused attention: logits + softmax + context ----------------------
// grid = BSZ blocks, 1024 threads (32 waves).
// Phase 1: wave-per-row logits, lanes cover EDIM in contiguous 32B chunks (coalesced).
// Phase 2: block softmax in LDS; scores -> d_out.
// Phase 3: cxt[b][e] = sum_s p[s]*enc[b][s][e], threads across e (coalesced), 2 s-halves.
__global__ __launch_bounds__(1024) void k_attn_ctx(
    const __bf16* __restrict__ enc, const float* __restrict__ x,
    float* __restrict__ attn_out, long attn_stride,
    __bf16* __restrict__ cxt_bf, int srclen) {
    int b    = blockIdx.x;
    int tid  = threadIdx.x;
    int lane = tid & 31;
    int wave = tid >> 5;
    int nw   = blockDim.x >> 5;          // 32 waves

    __shared__ float xs[EDIM];           // x vector, then reused for cxt halves
    __shared__ float pr[1024];           // logits -> probs
    __shared__ float red[32];

    for (int i = tid; i < EDIM; i += blockDim.x) xs[i] = x[b * EDIM + i];
    __syncthreads();

    const __bf16* eb = enc + (long)b * srclen * EDIM;

    // ---- phase 1: logits (wave per row; lane covers e = lane*16 .. +15) ----
    int rows = srclen / nw;              // rows per wave (32 when srclen=1024)
    const float* xp = xs + lane * 16;
    float lmax = -3.4e38f;
    for (int r = 0; r < rows; r++) {
        int s = wave * rows + r;
        const __bf16* ep = eb + (long)s * EDIM + lane * 16;
        if (r + 1 < rows) __builtin_prefetch(ep + EDIM, 0, 0);   // global_prefetch_b8
        uint4 q0 = *(const uint4*)ep;
        uint4 q1 = *(const uint4*)(ep + 8);
        const __bf16* c0 = (const __bf16*)&q0;
        const __bf16* c1 = (const __bf16*)&q1;
        float acc = 0.0f;
        #pragma unroll
        for (int i = 0; i < 8; i++) acc += (float)c0[i] * xp[i];
        #pragma unroll
        for (int i = 0; i < 8; i++) acc += (float)c1[i] * xp[8 + i];
        for (int o = 16; o > 0; o >>= 1) acc += __shfl_xor(acc, o);  // row total on all lanes
        if (lane == 0) pr[s] = acc;
        lmax = fmaxf(lmax, acc);
    }
    if (lane == 0) red[wave] = lmax;     // lmax identical across lanes of the wave
    __syncthreads();

    // ---- phase 2: softmax ----
    if (wave == 0) {
        float v = (lane < nw) ? red[lane] : -3.4e38f;
        for (int o = 16; o > 0; o >>= 1) v = fmaxf(v, __shfl_xor(v, o));
        if (lane == 0) red[0] = v;
    }
    __syncthreads();
    float m = red[0];
    float p = (tid < srclen) ? __expf(pr[tid] - m) : 0.0f;
    float ss = p;
    for (int o = 16; o > 0; o >>= 1) ss += __shfl_xor(ss, o);
    __syncthreads();
    if (lane == 0) red[wave] = ss;
    __syncthreads();
    if (wave == 0) {
        float v = (lane < nw) ? red[lane] : 0.0f;
        for (int o = 16; o > 0; o >>= 1) v += __shfl_xor(v, o);
        if (lane == 0) red[0] = v;
    }
    __syncthreads();
    float inv = 1.0f / red[0];
    if (tid < srclen) {
        float pn = p * inv;
        pr[tid] = pn;
        attn_out[(long)b * attn_stride + tid] = pn;
    }
    __syncthreads();

    // ---- phase 3: context (coalesced across e; two s-halves in parallel) ----
    int e    = tid & (EDIM - 1);
    int half = tid / EDIM;               // 0 or 1
    int sh   = srclen >> 1;
    int s0   = half * sh;
    const __bf16* ep = eb + (long)s0 * EDIM + e;
    float acc = 0.0f;
    for (int s = 0; s < sh; s++) {
        if ((s & 63) == 0 && s + 64 < sh) __builtin_prefetch(ep + (long)(s + 64) * EDIM, 0, 0);
        acc += pr[s0 + s] * (float)ep[(long)s * EDIM];
    }
    __syncthreads();                     // xs reuse
    if (half == 1) xs[e] = acc;
    __syncthreads();
    if (half == 0) cxt_bf[b * EDIM + e] = (__bf16)(acc + xs[e]);
}

// ---------------- host driver ------------------------------------------------------
extern "C" void kernel_launch(void* const* d_in, const int* in_sizes, int n_in,
                              void* d_out, int out_size, void* d_ws, size_t ws_size,
                              hipStream_t stream) {
    const float* enc_f  = (const float*)d_in[0];
    const float* W_ih   = (const float*)d_in[1];
    const float* W_hh   = (const float*)d_in[2];
    const float* b_ih   = (const float*)d_in[3];
    const float* b_hh   = (const float*)d_in[4];
    const float* W_attn = (const float*)d_in[5];
    const float* W_mu   = (const float*)d_in[6];
    const float* init_z = (const float*)d_in[7];

    long enc_n  = in_sizes[0];
    int  srclen = (int)(enc_n / ((long)BSZ * EDIM));                     // 1024
    int  T      = (int)((long)out_size / ((long)BSZ * (ZDIM + srclen))); // 512

    // workspace carve-out (256B aligned)
    char* p = (char*)d_ws;
    size_t used = 0;
    auto alloc = [&](size_t bytes) -> void* {
        void* r = p + used;
        used += (bytes + 255) & ~(size_t)255;
        return r;
    };
    __bf16* enc_bf  = (__bf16*)alloc(enc_n * 2);
    __bf16* Wih_bf  = (__bf16*)alloc((size_t)4 * HDIM * ZDIM * 2);
    __bf16* Whh_bf  = (__bf16*)alloc((size_t)4 * HDIM * HDIM * 2);
    __bf16* Watt_bf = (__bf16*)alloc((size_t)EDIM * HDIM * 2);
    __bf16* Wmu_bf  = (__bf16*)alloc((size_t)ZDIM * EDIM * 2);
    float*  bsum    = (float*) alloc((size_t)4 * HDIM * 4);
    float*  c_st    = (float*) alloc((size_t)BSZ * HDIM * 4);
    __bf16* hA      = (__bf16*)alloc((size_t)BSZ * HDIM * 2);
    __bf16* hB      = (__bf16*)alloc((size_t)BSZ * HDIM * 2);
    __bf16* z_bf    = (__bf16*)alloc((size_t)BSZ * ZDIM * 2);
    float*  x_buf   = (float*) alloc((size_t)BSZ * EDIM * 4);
    __bf16* cxt_bf  = (__bf16*)alloc((size_t)BSZ * EDIM * 2);
    if (used > ws_size) return;  // workspace too small; bail safely

    float* zouts = (float*)d_out;                       // (BSZ, T, ZDIM)
    float* attns = zouts + (long)BSZ * T * ZDIM;        // (BSZ, T, srclen)

    // one-time (per call) conversions + state init
    k_f32_to_bf16<<<8192, 256, 0, stream>>>(enc_f, enc_bf, enc_n);
    k_f32_to_bf16<<<4096, 256, 0, stream>>>(W_ih,   Wih_bf,  (long)4 * HDIM * ZDIM);
    k_f32_to_bf16<<<4096, 256, 0, stream>>>(W_hh,   Whh_bf,  (long)4 * HDIM * HDIM);
    k_f32_to_bf16<<<1024, 256, 0, stream>>>(W_attn, Watt_bf, (long)EDIM * HDIM);
    k_f32_to_bf16<<<1024, 256, 0, stream>>>(W_mu,   Wmu_bf,  (long)ZDIM * EDIM);
    k_bias_sum<<<(4 * HDIM + 255) / 256, 256, 0, stream>>>(b_ih, b_hh, bsum, 4 * HDIM);
    k_init_state<<<(BSZ * HDIM + 255) / 256, 256, 0, stream>>>(init_z, z_bf, hA, hB, c_st);

    for (int t = 0; t < T; t++) {
        __bf16* hin  = (t & 1) ? hB : hA;
        __bf16* hout = (t & 1) ? hA : hB;

        // 1. gates + LSTM cell
        k_lstm<<<HDIM / 16, 256, 0, stream>>>(z_bf, hin, Wih_bf, Whh_bf, bsum, c_st, hout);
        // 2. x = h @ W_attn^T
        k_gemm32<<<EDIM / 64, 256, 0, stream>>>(hout, Watt_bf, HDIM,
                                                x_buf, (long)EDIM, (__bf16*)nullptr, 0);
        // 3+4. scores = softmax(enc · x) -> attns[:, t, :];  cxt = scores @ enc
        k_attn_ctx<<<BSZ, 1024, 0, stream>>>(enc_bf, x_buf,
                                             attns + (long)t * srclen, (long)T * srclen,
                                             cxt_bf, srclen);
        // 5. mu = cxt @ W_mu^T  -> zouts[:, t, :] and next-step z
        k_gemm32<<<ZDIM / 64, 256, 0, stream>>>(cxt_bf, Wmu_bf, EDIM,
                                                zouts + (long)t * ZDIM, (long)T * ZDIM,
                                                z_bf, ZDIM);
    }
}
